// MoE_GRN_14396730376786
// MI455X (gfx1250) — compile-verified, matrix operand order checked by reading, not run on hardware
//
#include <hip/hip_runtime.h>
#include <hip/hip_bf16.h>
#include <math.h>

// ---------------------------------------------------------------------------
// MoE with GRN gating, MI455X (gfx1250, wave32, WMMA).
//
// Sizes: B=4096, IN=1024, HID=2048, OUT=1024, E=8, C=1000, TOPK=2
// fc2 is compute bound (AI ~2000 FLOP/B >> 23.3 TB/s crossover) ->
//   bf16 WMMA (v_wmma_f32_16x16x32_bf16) + top-2 routing (4x FLOP cut).
// CDNA5 paths: global_load_async_to_lds_b128 (ASYNCcnt) for bf16 A-tiles in
// fc2, ping-pong LDS double buffering (1 barrier/K-step), global_prefetch.
// ---------------------------------------------------------------------------

#define B_SZ   4096
#define IN_SZ  1024
#define HID_SZ 2048
#define OUT_SZ 1024
#define E_SZ   8
#define C_SZ   1000
#define HE     (HID_SZ * E_SZ)   // 16384
#define GRN_EPS 1e-6f

typedef __attribute__((ext_vector_type(4)))  __bf16 v4bf;
typedef __attribute__((ext_vector_type(8)))  __bf16 v8bf;
typedef __attribute__((ext_vector_type(16))) __bf16 v16bf;
typedef __attribute__((ext_vector_type(8)))  float  v8f;

#define WMMA_BF16(a, b, c) \
  __builtin_amdgcn_wmma_f32_16x16x32_bf16(false, (a), false, (b), (short)0, (c), false, false)

__device__ __forceinline__ v8f zero8() {
  v8f z;
#pragma unroll
  for (int i = 0; i < 8; ++i) z[i] = 0.0f;
  return z;
}

// Async DMA: global (bf16, 16B) -> LDS, tracked by ASYNCcnt (ISA 10.7/15.18).
// LDS operand = wave-relative LDS byte address = low 32 bits of the generic
// pointer (ISA 10.2).
__device__ __forceinline__ void async_copy_b128(const void* g, void* lds) {
  asm volatile("global_load_async_to_lds_b128 %0, %1, off"
               :: "v"((unsigned)(size_t)lds),
                  "v"((unsigned long long)(size_t)g)
               : "memory");
}

__device__ __forceinline__ void async_wait0() {
#if __has_builtin(__builtin_amdgcn_s_wait_asynccnt)
  __builtin_amdgcn_s_wait_asynccnt(0);
#else
  asm volatile("s_wait_asynccnt 0x0" ::: "memory");
#endif
}

// Load one 16x32 bf16 fragment from an LDS tile (rows padded to 40 elems).
// ISA 16-bit A/B layout: lane L holds row/col (L&15), K runs [kb..kb+7] and
// [kb+16..kb+23], kb = (L>>4)*8. Two ds_load_b128 per fragment.
__device__ __forceinline__ v16bf ldfrag(const __bf16 (*tile)[40], int base, int lane) {
  const __bf16* p = &tile[base + (lane & 15)][0];
  const int kb = (lane >> 4) << 3;
  v8bf lo = *(const v8bf*)(p + kb);
  v8bf hi = *(const v8bf*)(p + kb + 16);
  v16bf o;
#pragma unroll
  for (int i = 0; i < 8; ++i) { o[i] = lo[i]; o[i + 8] = hi[i]; }
  return o;
}

// ---------------------------------------------------------------------------
// Small kernels: init, gating, GRN statistics, softmax/top-k routing
// ---------------------------------------------------------------------------

__global__ void k_zero(float* gsum, int* cnt) {
  if (threadIdx.x == 0) *gsum = 0.0f;
  if (threadIdx.x < E_SZ) cnt[threadIdx.x] = 0;
}

__global__ void k_gate(const float* __restrict__ x, const float* __restrict__ Wg,
                       const float* __restrict__ bg, float* __restrict__ gl) {
  const int idx = blockIdx.x * blockDim.x + threadIdx.x;   // B*E threads
  const int row = idx >> 3, e = idx & 7;
  const float4* xp = (const float4*)(x + (size_t)row * IN_SZ);
  const float4* wp = (const float4*)(Wg + (size_t)e * IN_SZ);
  float acc = 0.0f;
#pragma unroll 8
  for (int i = 0; i < IN_SZ / 4; ++i) {
    float4 a = xp[i], b = wp[i];
    acc += a.x * b.x + a.y * b.y + a.z * b.z + a.w * b.w;
  }
  gl[idx] = acc + bg[e];
}

__global__ void k_rownorm(const float* __restrict__ gl, float* __restrict__ Gx) {
  const int row = blockIdx.x * blockDim.x + threadIdx.x;
  float s = 0.0f;
#pragma unroll
  for (int e = 0; e < E_SZ; ++e) { float v = gl[row * E_SZ + e]; s += v * v; }
  Gx[row] = sqrtf(s);
}

// Deterministic fixed-order reduction (no fp atomics -> replay-stable).
__global__ void k_reduce(const float* __restrict__ Gx, float* __restrict__ gsum) {
  __shared__ float sm[256];
  float s = 0.0f;
  for (int i = threadIdx.x; i < B_SZ; i += 256) s += Gx[i];
  sm[threadIdx.x] = s;
  __syncthreads();
#pragma unroll
  for (int off = 128; off > 0; off >>= 1) {
    if ((int)threadIdx.x < off) sm[threadIdx.x] += sm[threadIdx.x + off];
    __syncthreads();
  }
  if (threadIdx.x == 0) *gsum = sm[0];
}

__global__ void k_topk(const float* __restrict__ gl, const float* __restrict__ Gx,
                       const float* __restrict__ gsum,
                       const float* __restrict__ gamma, const float* __restrict__ beta,
                       float* __restrict__ tkp, int* __restrict__ cnt,
                       int* __restrict__ rows, int* __restrict__ slots) {
  const int row = blockIdx.x * blockDim.x + threadIdx.x;
  const float gmean = (*gsum) * (1.0f / (float)B_SZ);
  const float nx = Gx[row] / (gmean + GRN_EPS);
  float p[E_SZ];
  float mx = -3.0e38f;
#pragma unroll
  for (int e = 0; e < E_SZ; ++e) {
    float l = gamma[e] * (gl[row * E_SZ + e] * nx) + beta[e];
    p[e] = l;
    mx = fmaxf(mx, l);
  }
  float s = 0.0f;
#pragma unroll
  for (int e = 0; e < E_SZ; ++e) { p[e] = __expf(p[e] - mx); s += p[e]; }
  const float inv = 1.0f / s;
  int i0 = 0;
#pragma unroll
  for (int e = 1; e < E_SZ; ++e) if (p[e] > p[i0]) i0 = e;   // ties -> lowest idx
  int i1 = (i0 == 0) ? 1 : 0;
#pragma unroll
  for (int e = 0; e < E_SZ; ++e) if (e != i0 && p[e] > p[i1]) i1 = e;
  const int es[2] = { i0, i1 };
#pragma unroll
  for (int k = 0; k < 2; ++k) {
    const int e = es[k];
    tkp[row * 2 + k] = p[e] * inv;
    const int pos = atomicAdd(&cnt[e], 1);
    rows[e * B_SZ + pos]  = row;
    slots[e * B_SZ + pos] = row * 2 + k;   // slot-addressed -> deterministic output
  }
}

// ---------------------------------------------------------------------------
// fc1: h = relu(x @ W1.T + b1), stored bf16. 128x128x32 tile, ping-pong LDS.
// ---------------------------------------------------------------------------
__global__ __launch_bounds__(256) void k_fc1(const float* __restrict__ x,
                                             const float* __restrict__ W1,
                                             const float* __restrict__ b1,
                                             __bf16* __restrict__ h) {
  __shared__ __bf16 As[2][128][40];
  __shared__ __bf16 Bs[2][128][40];
  const int tid = threadIdx.x;
  const int m0 = blockIdx.x * 128, n0 = blockIdx.y * 128;
  const int lane = tid & 31, wave = tid >> 5;
  const int wm = wave >> 1, wn = wave & 1;

  auto stage = [&](int buf, int k0) {
#pragma unroll
    for (int ps = 0; ps < 4; ++ps) {
      const int r = (tid >> 3) + ps * 32;
      const int kq = (tid & 7) * 4;
      float4 a = *(const float4*)(x  + (size_t)(m0 + r) * IN_SZ + k0 + kq);
      float4 w = *(const float4*)(W1 + (size_t)(n0 + r) * IN_SZ + k0 + kq);
      v4bf ba; ba[0] = (__bf16)a.x; ba[1] = (__bf16)a.y; ba[2] = (__bf16)a.z; ba[3] = (__bf16)a.w;
      v4bf bw; bw[0] = (__bf16)w.x; bw[1] = (__bf16)w.y; bw[2] = (__bf16)w.z; bw[3] = (__bf16)w.w;
      *(v4bf*)(&As[buf][r][kq]) = ba;
      *(v4bf*)(&Bs[buf][r][kq]) = bw;
    }
  };

  v8f acc[2][4];
#pragma unroll
  for (int mt = 0; mt < 2; ++mt)
#pragma unroll
    for (int nt = 0; nt < 4; ++nt) acc[mt][nt] = zero8();

  stage(0, 0);
  __syncthreads();
  const int NT = IN_SZ / 32;
  for (int kt = 0; kt < NT; ++kt) {
    const int cur = kt & 1;
    if (kt + 1 < NT) stage(cur ^ 1, (kt + 1) * 32);
    v16bf af[2], bfv[4];
    af[0] = ldfrag(As[cur], wm * 32, lane);
    af[1] = ldfrag(As[cur], wm * 32 + 16, lane);
#pragma unroll
    for (int nt = 0; nt < 4; ++nt) bfv[nt] = ldfrag(Bs[cur], wn * 64 + nt * 16, lane);
#pragma unroll
    for (int mt = 0; mt < 2; ++mt)
#pragma unroll
      for (int nt = 0; nt < 4; ++nt) acc[mt][nt] = WMMA_BF16(af[mt], bfv[nt], acc[mt][nt]);
    __syncthreads();
  }

  const int lm = (lane >> 4) * 8, ln = lane & 15;   // C/D layout
#pragma unroll
  for (int mt = 0; mt < 2; ++mt)
#pragma unroll
    for (int nt = 0; nt < 4; ++nt) {
      const int col = n0 + wn * 64 + nt * 16 + ln;
      const float bias = b1[col];
#pragma unroll
      for (int r = 0; r < 8; ++r) {
        const int rowg = m0 + wm * 32 + mt * 16 + lm + r;
        float v = acc[mt][nt][r] + bias;
        h[(size_t)rowg * HE + col] = (__bf16)fmaxf(v, 0.0f);
      }
    }
}

// ---------------------------------------------------------------------------
// fc2 (routed): gather h rows for expert e, GEMM vs W2[e*OUT:(e+1)*OUT, :].
// A-tile (bf16) staged with global_load_async_to_lds_b128 (ASYNCcnt);
// B-tile fp32->bf16 via VALU (co-executes with XDL WMMA). Ping-pong LDS.
// ---------------------------------------------------------------------------
__global__ __launch_bounds__(256) void k_fc2(const __bf16* __restrict__ h,
                                             const float* __restrict__ W2,
                                             const float* __restrict__ b2,
                                             const int* __restrict__ cnt,
                                             const int* __restrict__ rows,
                                             const int* __restrict__ slots,
                                             float* __restrict__ buf2) {
  const int e = blockIdx.z;
  const int m0 = blockIdx.x * 128, n0 = blockIdx.y * 128;
  const int cE = cnt[e];
  if (m0 >= cE) return;                       // uniform early exit (empty tile)

  __shared__ int rIdx[128];
  __shared__ int sIdx[128];
  __shared__ __bf16 As[2][128][40];
  __shared__ __bf16 Bs[2][128][40];
  const int tid = threadIdx.x;
  if (tid < 128) {
    const int i = m0 + tid;
    rIdx[tid] = (i < cE) ? rows[e * B_SZ + i] : 0;
    sIdx[tid] = (i < cE) ? slots[e * B_SZ + i] : -1;
  }
  __syncthreads();

  const int lane = tid & 31, wave = tid >> 5;
  const int wm = wave >> 1, wn = wave & 1;

  auto stage = [&](int buf, int k0) {
    // A: async DMA of gathered bf16 rows, 2 x 16B per thread
#pragma unroll
    for (int c = 0; c < 2; ++c) {
      const int cid = tid + c * 256;
      const int r = cid >> 2, kc = (cid & 3) * 8;
      async_copy_b128(h + (size_t)rIdx[r] * HE + k0 + kc, &As[buf][r][kc]);
    }
    // B: W2 fp32 -> bf16 through VGPRs (overlaps with WMMA)
#pragma unroll
    for (int ps = 0; ps < 4; ++ps) {
      const int r = (tid >> 3) + ps * 32;
      const int kq = (tid & 7) * 4;
      const float* src = W2 + (size_t)(e * OUT_SZ + n0 + r) * HE + k0 + kq;
      float4 w = *(const float4*)src;
      if (ps == 0 && k0 + 64 < HE) __builtin_prefetch(src + 64, 0, 1);
      v4bf bw; bw[0] = (__bf16)w.x; bw[1] = (__bf16)w.y; bw[2] = (__bf16)w.z; bw[3] = (__bf16)w.w;
      *(v4bf*)(&Bs[buf][r][kq]) = bw;
    }
  };

  v8f acc[2][4];
#pragma unroll
  for (int mt = 0; mt < 2; ++mt)
#pragma unroll
    for (int nt = 0; nt < 4; ++nt) acc[mt][nt] = zero8();

  stage(0, 0);
  async_wait0();
  __syncthreads();
  const int NT = HE / 32;
  for (int kt = 0; kt < NT; ++kt) {
    const int cur = kt & 1;
    if (kt + 1 < NT) stage(cur ^ 1, (kt + 1) * 32);
    v16bf af[2], bfv[4];
    af[0] = ldfrag(As[cur], wm * 32, lane);
    af[1] = ldfrag(As[cur], wm * 32 + 16, lane);
#pragma unroll
    for (int nt = 0; nt < 4; ++nt) bfv[nt] = ldfrag(Bs[cur], wn * 64 + nt * 16, lane);
#pragma unroll
    for (int mt = 0; mt < 2; ++mt)
#pragma unroll
      for (int nt = 0; nt < 4; ++nt) acc[mt][nt] = WMMA_BF16(af[mt], bfv[nt], acc[mt][nt]);
    async_wait0();     // next tile's async A writes complete before barrier
    __syncthreads();
  }

  const int lm = (lane >> 4) * 8, ln = lane & 15;
#pragma unroll
  for (int mt = 0; mt < 2; ++mt)
#pragma unroll
    for (int nt = 0; nt < 4; ++nt) {
      const int col = n0 + wn * 64 + nt * 16 + ln;
      const float bias = b2[e * OUT_SZ + col];
#pragma unroll
      for (int r = 0; r < 8; ++r) {
        const int rl = wm * 32 + mt * 16 + lm + r;
        const int slot = sIdx[rl];
        if (slot >= 0) buf2[(size_t)slot * OUT_SZ + col] = acc[mt][nt][r] + bias;
      }
    }
}

// mix[b,:] = p0 * buf2[b*2+0,:] + p1 * buf2[b*2+1,:]
__global__ void k_combine(const float* __restrict__ buf2, const float* __restrict__ tkp,
                          float* __restrict__ mix) {
  const int idx = blockIdx.x * blockDim.x + threadIdx.x;   // B*OUT threads
  const int row = idx >> 10, col = idx & 1023;
  mix[idx] = tkp[row * 2 + 0] * buf2[(size_t)(row * 2 + 0) * OUT_SZ + col]
           + tkp[row * 2 + 1] * buf2[(size_t)(row * 2 + 1) * OUT_SZ + col];
}

// ---------------------------------------------------------------------------
// classifier: out = mix @ Wc.T + bc   (N = 1000, edge tiles guarded)
// ---------------------------------------------------------------------------
__global__ __launch_bounds__(256) void k_cls(const float* __restrict__ mix,
                                             const float* __restrict__ Wc,
                                             const float* __restrict__ bc,
                                             float* __restrict__ out) {
  __shared__ __bf16 As[2][128][40];
  __shared__ __bf16 Bs[2][128][40];
  const int tid = threadIdx.x;
  const int m0 = blockIdx.x * 128, n0 = blockIdx.y * 128;
  const int lane = tid & 31, wave = tid >> 5;
  const int wm = wave >> 1, wn = wave & 1;

  auto stage = [&](int buf, int k0) {
#pragma unroll
    for (int ps = 0; ps < 4; ++ps) {
      const int r = (tid >> 3) + ps * 32;
      const int kq = (tid & 7) * 4;
      float4 a = *(const float4*)(mix + (size_t)(m0 + r) * OUT_SZ + k0 + kq);
      v4bf ba; ba[0] = (__bf16)a.x; ba[1] = (__bf16)a.y; ba[2] = (__bf16)a.z; ba[3] = (__bf16)a.w;
      *(v4bf*)(&As[buf][r][kq]) = ba;
      const int n = n0 + r;
      v4bf bw;
      if (n < C_SZ) {
        float4 w = *(const float4*)(Wc + (size_t)n * OUT_SZ + k0 + kq);
        bw[0] = (__bf16)w.x; bw[1] = (__bf16)w.y; bw[2] = (__bf16)w.z; bw[3] = (__bf16)w.w;
      } else {
        bw[0] = (__bf16)0.0f; bw[1] = (__bf16)0.0f; bw[2] = (__bf16)0.0f; bw[3] = (__bf16)0.0f;
      }
      *(v4bf*)(&Bs[buf][r][kq]) = bw;
    }
  };

  v8f acc[2][4];
#pragma unroll
  for (int mt = 0; mt < 2; ++mt)
#pragma unroll
    for (int nt = 0; nt < 4; ++nt) acc[mt][nt] = zero8();

  stage(0, 0);
  __syncthreads();
  const int NT = OUT_SZ / 32;
  for (int kt = 0; kt < NT; ++kt) {
    const int cur = kt & 1;
    if (kt + 1 < NT) stage(cur ^ 1, (kt + 1) * 32);
    v16bf af[2], bfv[4];
    af[0] = ldfrag(As[cur], wm * 32, lane);
    af[1] = ldfrag(As[cur], wm * 32 + 16, lane);
#pragma unroll
    for (int nt = 0; nt < 4; ++nt) bfv[nt] = ldfrag(Bs[cur], wn * 64 + nt * 16, lane);
#pragma unroll
    for (int mt = 0; mt < 2; ++mt)
#pragma unroll
      for (int nt = 0; nt < 4; ++nt) acc[mt][nt] = WMMA_BF16(af[mt], bfv[nt], acc[mt][nt]);
    __syncthreads();
  }

  const int lm = (lane >> 4) * 8, ln = lane & 15;
#pragma unroll
  for (int mt = 0; mt < 2; ++mt)
#pragma unroll
    for (int nt = 0; nt < 4; ++nt) {
      const int col = n0 + wn * 64 + nt * 16 + ln;
      if (col >= C_SZ) continue;
      const float bias = bc[col];
#pragma unroll
      for (int r = 0; r < 8; ++r) {
        const int rowg = m0 + wm * 32 + mt * 16 + lm + r;
        out[(size_t)rowg * C_SZ + col] = acc[mt][nt][r] + bias;
      }
    }
}

// ---------------------------------------------------------------------------
extern "C" void kernel_launch(void* const* d_in, const int* in_sizes, int n_in,
                              void* d_out, int out_size, void* d_ws, size_t ws_size,
                              hipStream_t stream) {
  (void)in_sizes; (void)n_in; (void)out_size; (void)ws_size;
  const float* x     = (const float*)d_in[0];
  const float* Wg    = (const float*)d_in[1];
  const float* bg    = (const float*)d_in[2];
  const float* gamma = (const float*)d_in[3];
  const float* beta  = (const float*)d_in[4];
  const float* W1    = (const float*)d_in[5];
  const float* b1    = (const float*)d_in[6];
  const float* W2    = (const float*)d_in[7];
  const float* b2    = (const float*)d_in[8];
  const float* Wc    = (const float*)d_in[9];
  const float* bc    = (const float*)d_in[10];
  float* out = (float*)d_out;

  char* ws = (char*)d_ws;
  size_t off = 0;
  auto take = [&](size_t bytes) -> char* {
    char* p = ws + off;
    off = (off + bytes + 255) & ~(size_t)255;
    return p;
  };
  float*  gl    = (float*)take((size_t)B_SZ * E_SZ * 4);      // gate logits
  float*  Gx    = (float*)take((size_t)B_SZ * 4);             // per-row norm
  float*  gsum  = (float*)take(4);                            // batch sum of norms
  float*  tkp   = (float*)take((size_t)B_SZ * 2 * 4);         // top-2 probs
  int*    cnt   = (int*)  take((size_t)E_SZ * 4);             // per-expert counts
  int*    rowsL = (int*)  take((size_t)E_SZ * B_SZ * 4);      // per-expert row lists
  int*    slotL = (int*)  take((size_t)E_SZ * B_SZ * 4);      // per-expert slot lists
  __bf16* h     = (__bf16*)take((size_t)B_SZ * HE * 2);       // 128 MB, bf16 activations
  float*  buf2  = (float*)take((size_t)B_SZ * 2 * OUT_SZ * 4);// 32 MB, per-assignment out
  float*  mixb  = (float*)take((size_t)B_SZ * OUT_SZ * 4);    // 16 MB, mixed expert out

  k_zero<<<1, 32, 0, stream>>>(gsum, cnt);
  k_gate<<<(B_SZ * E_SZ) / 256, 256, 0, stream>>>(x, Wg, bg, gl);
  k_rownorm<<<B_SZ / 256, 256, 0, stream>>>(gl, Gx);
  k_reduce<<<1, 256, 0, stream>>>(Gx, gsum);
  k_topk<<<B_SZ / 256, 256, 0, stream>>>(gl, Gx, gsum, gamma, beta, tkp, cnt, rowsL, slotL);
  k_fc1<<<dim3(B_SZ / 128, HE / 128), 256, 0, stream>>>(x, W1, b1, h);
  k_fc2<<<dim3(B_SZ / 128, OUT_SZ / 128, E_SZ), 256, 0, stream>>>(h, W2, b2, cnt, rowsL, slotL, buf2);
  k_combine<<<(B_SZ * OUT_SZ) / 256, 256, 0, stream>>>(buf2, tkp, mixb);
  k_cls<<<dim3(B_SZ / 128, (C_SZ + 127) / 128), 256, 0, stream>>>(mixb, Wc, bc, out);
}
